// GatedDeltaMemory_87883620811458
// MI455X (gfx1250) — matrix operand order repaired and checked by reading, not compiled, not run on hardware
//
#include <hip/hip_runtime.h>
#include <math.h>

// ---------------------------------------------------------------------------
// Types for CDNA5 WMMA (wave32, 16x16x32 bf16 -> f32)
// ---------------------------------------------------------------------------
typedef __bf16 bf16;
typedef bf16  v16bf __attribute__((ext_vector_type(16)));
typedef bf16  v8bf  __attribute__((ext_vector_type(8)));
typedef bf16  v4bf  __attribute__((ext_vector_type(4)));
typedef float v8f   __attribute__((ext_vector_type(8)));

#define D_MODEL 768
#define D_FF    3072
#define N_HEAD  12
#define D_HEAD  64
#define BT      256      // B*T
#define N_LAT   16
#define R_ROWS  4096     // BT*N_LAT  (also L_LANES*T_SEQ)
#define KV_ROWS 8192     // BT*2*N_LAT
#define T_SEQ   128
#define L_LANES 32       // B*N_LAT

static __device__ __forceinline__ float sigmoidf_(float x){ return 1.f/(1.f+__expf(-x)); }
static __device__ __forceinline__ float siluf_(float x){ return x*sigmoidf_(x); }
static __device__ __forceinline__ float geluf_(float x){
  float x3 = x*x*x;
  return 0.5f*x*(1.f+tanhf(0.7978845608028654f*(x+0.044715f*x3)));
}
static __device__ __forceinline__ float softplusf_(float x){
  return (x > 20.f) ? x : log1pf(__expf(x));
}

// ---------------------------------------------------------------------------
// One-time per launch: W[K,N] fp32 -> Wt[N,K] bf16 (transposed weight arena).
// ---------------------------------------------------------------------------
__global__ __launch_bounds__(256) void transpose_cvt_kernel(
    const float* __restrict__ W, bf16* __restrict__ Wt, int K, int N)
{
  size_t idx = (size_t)blockIdx.x*256 + threadIdx.x;   // over Wt, k fastest
  if (idx >= (size_t)K*N) return;
  int k = (int)(idx % K);
  int n = (int)(idx / K);
  Wt[idx] = (bf16)W[(size_t)k*N + n];
}

// ---------------------------------------------------------------------------
// GEMM: C[M,N] = A[M,K] @ W[K,N]; A fp32, W pre-transposed bf16 Wt[N,K].
// Block: 256 threads = 8 waves; tile 128x128; K-step 32; double-buffered LDS.
// B tiles staged with CDNA5 async LDS DMA (global_load_async_to_lds_b128,
// ASYNCcnt-fenced) when the N-tile is full; guarded sync path otherwise.
// Fragment layouts (per ISA 7.12.2):
//   A lane (m=lane&15, half=lane>>4): i<8 -> K=8*half+i ; i>=8 -> K=16+8*half+(i-8)
//   B lane (n=lane&15, half):         i<8 -> K=8*half+i ; i>=8 -> K=16+8*half+(i-8)
//   C/D:  vgpr j -> m=j+8*half ; n=lane&15
// Requires M % 128 == 0 and K % 32 == 0 (true at all call sites); N guarded.
// ---------------------------------------------------------------------------
#define LDSA 40
#define LDSB 40

__global__ __launch_bounds__(256) void gemm_bf16_kernel(
    const float* __restrict__ A, const bf16* __restrict__ Wt,
    float* __restrict__ C, int M, int N, int K)
{
  __shared__ bf16 As[2][128*LDSA];
  __shared__ bf16 Bs[2][128*LDSB];
  const int tid  = threadIdx.x;
  const int wave = tid >> 5;
  const int lane = tid & 31;
  const int ml   = lane & 15;
  const int hf   = lane >> 4;
  const int m0   = blockIdx.x * 128;
  const int n0   = blockIdx.y * 128;
  const bool fullN = (n0 + 128) <= N;

  v8f acc[8];
  #pragma unroll
  for (int t = 0; t < 8; ++t)
    #pragma unroll
    for (int j = 0; j < 8; ++j) acc[t][j] = 0.f;

  float4 fa[4];
  auto loadA = [&](int k0) {
    #pragma unroll
    for (int i = 0; i < 4; ++i) {
      int c  = tid + i*256;            // 1024 float4 chunks: 128 rows x 8
      int am = c >> 3;
      int ak = (c & 7) << 2;
      fa[i] = *(const float4*)(A + (size_t)(m0+am)*K + k0 + ak);
    }
  };
  auto storeA = [&](int buf) {
    #pragma unroll
    for (int i = 0; i < 4; ++i) {
      int c  = tid + i*256;
      int am = c >> 3;
      int ak = (c & 7) << 2;
      v4bf b4; b4[0]=(bf16)fa[i].x; b4[1]=(bf16)fa[i].y;
               b4[2]=(bf16)fa[i].z; b4[3]=(bf16)fa[i].w;
      *(v4bf*)(&As[buf][am*LDSA + ak]) = b4;
    }
  };
  auto stageB = [&](int k0, int buf) {
    if (fullN) {
      // async DMA: 512 x 16B chunks (128 rows x 4), 2 per thread
      #pragma unroll
      for (int i = 0; i < 2; ++i) {
        int c  = tid + i*256;
        int n  = c >> 2;
        int kc = (c & 3) * 8;
        unsigned int lds = (unsigned int)(unsigned long long)
                           (&Bs[buf][n*LDSB + kc]);
        const bf16* gp = Wt + (size_t)(n0 + n)*K + k0 + kc;
        asm volatile("global_load_async_to_lds_b128 %0, %1, off"
                     :: "v"(lds), "v"(gp) : "memory");
      }
    } else {
      #pragma unroll
      for (int i = 0; i < 2; ++i) {
        int c  = tid + i*256;
        int n  = c >> 2;
        int kc = (c & 3) * 8;
        v8bf val;
        if (n0 + n < N) {
          val = *(const v8bf*)(Wt + (size_t)(n0 + n)*K + k0 + kc);
        } else {
          #pragma unroll
          for (int j = 0; j < 8; ++j) val[j] = (bf16)0.f;
        }
        *(v8bf*)(&Bs[buf][n*LDSB + kc]) = val;
      }
    }
  };

  // prologue: stage tile 0 into buffer 0
  loadA(0);
  storeA(0);
  stageB(0, 0);

  int cur = 0;
  for (int k0 = 0; k0 < K; k0 += 32) {
    const bool nxt = (k0 + 32) < K;
    if (fullN) asm volatile("s_wait_asynccnt 0x0" ::: "memory");
    __syncthreads();
    // issue next tile's transfers; they overlap this tile's WMMAs
    if (nxt) {
      loadA(k0 + 32);
      stageB(k0 + 32, cur ^ 1);
      if (k0 + 64 < K)
        __builtin_prefetch(A + (size_t)(m0 + (tid >> 1))*K + k0 + 64, 0, 0);
    }
    // compute: this wave owns rows wave*16..wave*16+15, all 8 n-tiles
    v8bf alo = *(const v8bf*)(&As[cur][(wave*16 + ml)*LDSA + 8*hf]);
    v8bf ahi = *(const v8bf*)(&As[cur][(wave*16 + ml)*LDSA + 16 + 8*hf]);
    v16bf af;
    #pragma unroll
    for (int j = 0; j < 8; ++j) { af[j]=alo[j]; af[8+j]=ahi[j]; }
    #pragma unroll
    for (int nt = 0; nt < 8; ++nt) {
      v8bf b0 = *(const v8bf*)(&Bs[cur][(nt*16 + ml)*LDSB + 8*hf]);
      v8bf b1 = *(const v8bf*)(&Bs[cur][(nt*16 + ml)*LDSB + 16 + 8*hf]);
      v16bf bfr;
      #pragma unroll
      for (int j = 0; j < 8; ++j) { bfr[j]=b0[j]; bfr[8+j]=b1[j]; }
      acc[nt] = __builtin_amdgcn_wmma_f32_16x16x32_bf16(
                  false, af, false, bfr, (short)0, acc[nt], false, false);
    }
    if (nxt) storeA(cur ^ 1);
    cur ^= 1;
  }

  // store C per D-matrix layout
  #pragma unroll
  for (int nt = 0; nt < 8; ++nt) {
    int col = n0 + nt*16 + ml;
    if (col < N) {
      #pragma unroll
      for (int j = 0; j < 8; ++j) {
        int row = m0 + wave*16 + j + 8*hf;
        C[(size_t)row*N + col] = acc[nt][j];
      }
    }
  }
}

// ---------------------------------------------------------------------------
// Fused small attention: one wave per (group, head).
// Q group = 16 rows, K/V group = Nk (16 or 32) rows, dh=64, d=768.
// ---------------------------------------------------------------------------
__global__ __launch_bounds__(32) void attn_kernel(
    const float* __restrict__ Qp, const float* __restrict__ Kp,
    const float* __restrict__ Vp, float* __restrict__ O,
    int Nk, float scale)
{
  __shared__ bf16 probs[16*32];
  const int h    = blockIdx.x % N_HEAD;
  const int g    = blockIdx.x / N_HEAD;
  const int lane = threadIdx.x;
  const int ml   = lane & 15;
  const int hf   = lane >> 4;
  const int qbase  = g * 16;
  const int kvbase = g * Nk;
  const int col0   = h * D_HEAD;
  const bool two   = (Nk == 32);

  v8f sc[2];
  #pragma unroll
  for (int t = 0; t < 2; ++t)
    #pragma unroll
    for (int j = 0; j < 8; ++j) sc[t][j] = 0.f;

  #pragma unroll
  for (int kc = 0; kc < 2; ++kc) {
    v16bf af;
    #pragma unroll
    for (int i = 0; i < 16; ++i) {
      int kk = (i < 8) ? (8*hf + i) : (16 + 8*hf + (i-8));
      af[i] = (bf16)Qp[(size_t)(qbase+ml)*D_MODEL + col0 + kc*32 + kk];
    }
    for (int t = 0; t < (two ? 2 : 1); ++t) {
      v16bf bfr;
      #pragma unroll
      for (int i = 0; i < 16; ++i) {
        int kk = (i < 8) ? (8*hf + i) : (16 + 8*hf + (i-8));
        bfr[i] = (bf16)Kp[(size_t)(kvbase + t*16 + ml)*D_MODEL + col0 + kc*32 + kk];
      }
      sc[t] = __builtin_amdgcn_wmma_f32_16x16x32_bf16(
                false, af, false, bfr, (short)0, sc[t], false, false);
    }
  }

  v8f p0, p1;
  #pragma unroll
  for (int j = 0; j < 8; ++j) {
    float s0 = sc[0][j]*scale;
    float s1 = two ? sc[1][j]*scale : -1e30f;
    float mx = fmaxf(s0, s1);
    #pragma unroll
    for (int msk = 1; msk < 16; msk <<= 1) mx = fmaxf(mx, __shfl_xor(mx, msk, 32));
    float e0 = __expf(s0 - mx);
    float e1 = two ? __expf(s1 - mx) : 0.f;
    float sm = e0 + e1;
    #pragma unroll
    for (int msk = 1; msk < 16; msk <<= 1) sm += __shfl_xor(sm, msk, 32);
    float r = 1.f / sm;
    p0[j] = e0*r; p1[j] = e1*r;
  }

  if (!two) for (int i = lane; i < 512; i += 32) probs[i] = (bf16)0.f;
  #pragma unroll
  for (int j = 0; j < 8; ++j) probs[(j + 8*hf)*32 + ml] = (bf16)p0[j];
  if (two) {
    #pragma unroll
    for (int j = 0; j < 8; ++j) probs[(j + 8*hf)*32 + 16 + ml] = (bf16)p1[j];
  }

  v8bf plo = *(const v8bf*)(&probs[ml*32 + 8*hf]);
  v8bf phi = *(const v8bf*)(&probs[ml*32 + 16 + 8*hf]);
  v16bf af2;
  #pragma unroll
  for (int j = 0; j < 8; ++j) { af2[j]=plo[j]; af2[8+j]=phi[j]; }
  #pragma unroll
  for (int nt = 0; nt < 4; ++nt) {
    v16bf bfr;
    #pragma unroll
    for (int i = 0; i < 16; ++i) {
      int key = (i < 8) ? (8*hf + i) : (16 + 8*hf + (i-8));
      float v = (key < Nk) ? Vp[(size_t)(kvbase+key)*D_MODEL + col0 + nt*16 + ml] : 0.f;
      bfr[i] = (bf16)v;
    }
    v8f o;
    #pragma unroll
    for (int j = 0; j < 8; ++j) o[j] = 0.f;
    o = __builtin_amdgcn_wmma_f32_16x16x32_bf16(
          false, af2, false, bfr, (short)0, o, false, false);
    #pragma unroll
    for (int j = 0; j < 8; ++j)
      O[(size_t)(qbase + j + 8*hf)*D_MODEL + col0 + nt*16 + ml] = o[j];
  }
}

// ---------------------------------------------------------------------------
// GDN scan: block = 64 threads, one (lane l, head h) per block.
// ---------------------------------------------------------------------------
__global__ __launch_bounds__(64) void gdn_scan_kernel(
    const float* __restrict__ q, const float* __restrict__ k,
    const float* __restrict__ v, const float* __restrict__ beta,
    const float* __restrict__ g, float* __restrict__ o)
{
  __shared__ float kk[64], qq[64];
  const int l  = blockIdx.x / N_HEAD;
  const int h  = blockIdx.x % N_HEAD;
  const int vc = threadIdx.x;
  const int col0 = h * D_HEAD;
  float S[64];
  #pragma unroll
  for (int i = 0; i < 64; ++i) S[i] = 0.f;

  for (int t = 0; t < T_SEQ; ++t) {
    size_t row = (size_t)l*T_SEQ + t;
    __syncthreads();
    kk[vc] = k[row*D_MODEL + col0 + vc];
    qq[vc] = q[row*D_MODEL + col0 + vc];
    __syncthreads();
    float decay = __expf(g[row*N_HEAD + h]);
    float bt    = beta[row*N_HEAD + h];
    float vt    = v[row*D_MODEL + col0 + vc];
    float mem = 0.f;
    #pragma unroll 8
    for (int i = 0; i < 64; ++i) mem += kk[i]*S[i];
    mem *= decay;
    float delta = (vt - mem)*bt;
    float ot = 0.f;
    #pragma unroll 8
    for (int i = 0; i < 64; ++i) {
      float s2 = S[i]*decay + kk[i]*delta;
      S[i] = s2;
      ot += qq[i]*s2;
    }
    o[row*D_MODEL + col0 + vc] = ot;
  }
}

// ---------------------------------------------------------------------------
// Elementwise / norm kernels
// ---------------------------------------------------------------------------
__global__ __launch_bounds__(256) void rmsnorm768_kernel(
    const float* __restrict__ x, const float* __restrict__ w,
    float* __restrict__ y)
{
  __shared__ float red[8];
  const int row = blockIdx.x;
  const int tid = threadIdx.x;
  const int wid = tid >> 5, lane = tid & 31;
  const float* xr = x + (size_t)row*D_MODEL;
  float v0 = xr[tid], v1 = xr[tid+256], v2 = xr[tid+512];
  float s = v0*v0 + v1*v1 + v2*v2;
  #pragma unroll
  for (int m = 1; m < 32; m <<= 1) s += __shfl_xor(s, m, 32);
  if (lane == 0) red[wid] = s;
  __syncthreads();
  float tot = 0.f;
  #pragma unroll
  for (int i = 0; i < 8; ++i) tot += red[i];
  float rr = rsqrtf(tot*(1.f/768.f) + 1e-5f);
  float* yr = y + (size_t)row*D_MODEL;
  yr[tid]     = v0*rr*w[tid];
  yr[tid+256] = v1*rr*w[tid+256];
  yr[tid+512] = v2*rr*w[tid+512];
}

__global__ __launch_bounds__(256) void l2norm_kernel(float* __restrict__ x, float scale, int nslices)
{
  const int wid = threadIdx.x >> 5, lane = threadIdx.x & 31;
  const int s = blockIdx.x*8 + wid;
  if (s >= nslices) return;
  float* p = x + (size_t)(s/N_HEAD)*D_MODEL + (s%N_HEAD)*D_HEAD;
  float a = p[lane], b = p[lane+32];
  float ss = a*a + b*b;
  #pragma unroll
  for (int m = 1; m < 32; m <<= 1) ss += __shfl_xor(ss, m, 32);
  float r = rsqrtf(ss + 1e-6f) * scale;
  p[lane]    = a*r;
  p[lane+32] = b*r;
}

__global__ __launch_bounds__(256) void gated_norm_kernel(
    float* __restrict__ o, const float* __restrict__ gate,
    const float* __restrict__ nw, int nslices)
{
  const int wid = threadIdx.x >> 5, lane = threadIdx.x & 31;
  const int s = blockIdx.x*8 + wid;
  if (s >= nslices) return;
  size_t base = (size_t)(s/N_HEAD)*D_MODEL + (s%N_HEAD)*D_HEAD;
  float a = o[base+lane], b = o[base+lane+32];
  float ss = a*a + b*b;
  #pragma unroll
  for (int m = 1; m < 32; m <<= 1) ss += __shfl_xor(ss, m, 32);
  float rr = rsqrtf(ss*(1.f/64.f) + 1e-5f);
  float ga = gate[base+lane], gb = gate[base+lane+32];
  o[base+lane]    = a*rr*nw[lane]   * siluf_(ga);
  o[base+lane+32] = b*rr*nw[lane+32]* siluf_(gb);
}

__global__ __launch_bounds__(256) void conv_silu_kernel(
    const float* __restrict__ x, const float* __restrict__ w,
    float* __restrict__ y)
{
  size_t idx = (size_t)blockIdx.x*256 + threadIdx.x;
  if (idx >= (size_t)R_ROWS*D_MODEL) return;
  int c  = idx % D_MODEL;
  int rt = idx / D_MODEL;
  int t  = rt % T_SEQ;
  int l  = rt / T_SEQ;
  float acc = 0.f;
  #pragma unroll
  for (int j = 0; j < 4; ++j) {
    int tt = t - 3 + j;
    if (tt >= 0) acc += x[((size_t)l*T_SEQ + tt)*D_MODEL + c] * w[c*4 + j];
  }
  y[idx] = siluf_(acc);
}

__global__ __launch_bounds__(256) void beta_g_kernel(
    const float* __restrict__ bpre, const float* __restrict__ gpre,
    const float* __restrict__ A_log, const float* __restrict__ dt_bias,
    float* __restrict__ beta, float* __restrict__ g)
{
  int idx = blockIdx.x*256 + threadIdx.x;
  if (idx >= R_ROWS*N_HEAD) return;
  int h = idx % N_HEAD;
  beta[idx] = sigmoidf_(bpre[idx]);
  g[idx]    = -__expf(A_log[h]) * softplusf_(gpre[idx] + dt_bias[h]);
}

__global__ __launch_bounds__(256) void gelu_kernel(float* __restrict__ x, size_t n)
{
  size_t idx = (size_t)blockIdx.x*256 + threadIdx.x;
  if (idx < n) x[idx] = geluf_(x[idx]);
}

__global__ __launch_bounds__(256) void add_kernel(
    const float* __restrict__ a, const float* __restrict__ b,
    float* __restrict__ y, size_t n)
{
  size_t idx = (size_t)blockIdx.x*256 + threadIdx.x;
  if (idx < n) y[idx] = a[idx] + b[idx];
}

__global__ __launch_bounds__(256) void concat_kernel(
    const float* __restrict__ p, const float* __restrict__ pa,
    float* __restrict__ y)
{
  int r = blockIdx.x;            // bt*32 + i
  int bt = r >> 5, i = r & 31;
  const float* src = (i < 16) ? (p  + (size_t)(bt*16 + i)*D_MODEL)
                              : (pa + (size_t)(bt*16 + (i-16))*D_MODEL);
  float* dst = y + (size_t)r*D_MODEL;
  for (int c = threadIdx.x; c < D_MODEL; c += 256) dst[c] = src[c];
}

__global__ __launch_bounds__(256) void broadcast_kernel(
    const float* __restrict__ lat, float* __restrict__ y)
{
  int r = blockIdx.x;
  const float* src = lat + (size_t)(r & 15)*D_MODEL;
  float* dst = y + (size_t)r*D_MODEL;
  for (int c = threadIdx.x; c < D_MODEL; c += 256) dst[c] = src[c];
}

__global__ __launch_bounds__(256) void perm_to_laned_kernel(
    const float* __restrict__ x, float* __restrict__ y)
{
  int r = blockIdx.x;            // laned row: (b*16+n)*128 + t
  int b = r >> 11, rem = r & 2047;
  int n = rem >> 7, t = rem & 127;
  const float* xs = x + (size_t)((b*T_SEQ + t)*N_LAT + n)*D_MODEL;
  float* yd = y + (size_t)r*D_MODEL;
  for (int c = threadIdx.x; c < D_MODEL; c += 256) yd[c] = xs[c];
}

__global__ __launch_bounds__(256) void perm_from_laned_add_kernel(
    float* __restrict__ state, const float* __restrict__ e)
{
  int r = blockIdx.x;            // btn row: (b*128+t)*16 + n
  int b = r >> 11, rem = r & 2047;
  int t = rem >> 4, n = rem & 15;
  const float* es = e + (size_t)((b*N_LAT + n)*T_SEQ + t)*D_MODEL;
  float* sd = state + (size_t)r*D_MODEL;
  for (int c = threadIdx.x; c < D_MODEL; c += 256) sd[c] += es[c];
}

// ---------------------------------------------------------------------------
// Host orchestration
// ---------------------------------------------------------------------------
extern "C" void kernel_launch(void* const* d_in, const int* in_sizes, int n_in,
                              void* d_out, int out_size, void* d_ws, size_t ws_size,
                              hipStream_t stream)
{
  (void)in_sizes; (void)n_in; (void)ws_size;
  const float* precepts = (const float*)d_in[0];
  const float* prev_act = (const float*)d_in[1];

  // ---- params, flattened in JAX pytree order (dict keys sorted per level)
  int pi = 2;
  auto nxt = [&]() { return (const float*)d_in[pi++]; };
  struct GdnP { const float *A_log,*ck,*cq,*cv,*dt_bias,*norm_w,*wa,*wb,*wg,*wk,*wo,*wq,*wv; };
  struct LayerP { const float *a_wk,*a_wo,*a_wq,*a_wv,*m_w1,*m_w2,*norm1,*norm2; };
  struct BlockP { GdnP gdn; const float* gdn_norm; LayerP lay[2]; };
  BlockP blk[4];
  for (int b = 0; b < 4; ++b) {
    GdnP& g = blk[b].gdn;
    g.A_log=nxt(); g.ck=nxt(); g.cq=nxt(); g.cv=nxt(); g.dt_bias=nxt();
    g.norm_w=nxt(); g.wa=nxt(); g.wb=nxt(); g.wg=nxt(); g.wk=nxt();
    g.wo=nxt(); g.wq=nxt(); g.wv=nxt();
    blk[b].gdn_norm = nxt();
    for (int l = 0; l < 2; ++l) {
      LayerP& L = blk[b].lay[l];
      L.a_wk=nxt(); L.a_wo=nxt(); L.a_wq=nxt(); L.a_wv=nxt();
      L.m_w1=nxt(); L.m_w2=nxt(); L.norm1=nxt(); L.norm2=nxt();
    }
  }
  const float* f_wk = nxt(); const float* f_wo = nxt();
  const float* f_wq = nxt(); const float* f_wv = nxt();
  const float* f_w1 = nxt(); const float* f_w2 = nxt();
  const float* f_ffn_norm = nxt();
  const float* f_latents  = nxt();
  const float* f_norm_kv  = nxt();

  // ---- bf16 transposed weight arena at the front of ws
  bf16* arena = (bf16*)d_ws;
  size_t apos = 0;
  auto cvtW = [&](const float* W, int K, int N) -> const bf16* {
    bf16* dst = arena + apos;
    size_t tot = (size_t)K * N;
    apos += tot;
    transpose_cvt_kernel<<<(int)((tot + 255)/256), 256, 0, stream>>>(W, dst, K, N);
    return dst;
  };
  struct GdnW  { const bf16 *wq,*wk,*wv,*wg,*wo,*wb,*wa; };
  struct LayW  { const bf16 *wq,*wk,*wv,*wo,*w1,*w2; };
  struct BlkW  { GdnW gdn; LayW lay[2]; };
  BlkW bw[4];
  for (int b = 0; b < 4; ++b) {
    for (int l = 0; l < 2; ++l) {
      const LayerP& L = blk[b].lay[l];
      bw[b].lay[l].wq = cvtW(L.a_wq, D_MODEL, D_MODEL);
      bw[b].lay[l].wk = cvtW(L.a_wk, D_MODEL, D_MODEL);
      bw[b].lay[l].wv = cvtW(L.a_wv, D_MODEL, D_MODEL);
      bw[b].lay[l].wo = cvtW(L.a_wo, D_MODEL, D_MODEL);
      bw[b].lay[l].w1 = cvtW(L.m_w1, D_MODEL, D_FF);
      bw[b].lay[l].w2 = cvtW(L.m_w2, D_FF, D_MODEL);
    }
    const GdnP& G = blk[b].gdn;
    bw[b].gdn.wq = cvtW(G.wq, D_MODEL, D_MODEL);
    bw[b].gdn.wk = cvtW(G.wk, D_MODEL, D_MODEL);
    bw[b].gdn.wv = cvtW(G.wv, D_MODEL, D_MODEL);
    bw[b].gdn.wg = cvtW(G.wg, D_MODEL, D_MODEL);
    bw[b].gdn.wo = cvtW(G.wo, D_MODEL, D_MODEL);
    bw[b].gdn.wb = cvtW(G.wb, D_MODEL, N_HEAD);
    bw[b].gdn.wa = cvtW(G.wa, D_MODEL, N_HEAD);
  }
  const bf16* tf_wq = cvtW(f_wq, D_MODEL, D_MODEL);
  const bf16* tf_wk = cvtW(f_wk, D_MODEL, D_MODEL);
  const bf16* tf_wv = cvtW(f_wv, D_MODEL, D_MODEL);
  const bf16* tf_wo = cvtW(f_wo, D_MODEL, D_MODEL);
  const bf16* tf_w1 = cvtW(f_w1, D_MODEL, D_FF);
  const bf16* tf_w2 = cvtW(f_w2, D_FF, D_MODEL);

  // ---- fp32 activation workspace after the arena (align to 256B)
  size_t abytes = (apos*sizeof(bf16) + 255) & ~(size_t)255;
  float* ws = (float*)((char*)d_ws + abytes);
  const size_t RD  = (size_t)R_ROWS * D_MODEL;
  const size_t KVD = (size_t)KV_ROWS * D_MODEL;
  float* W0 = ws;            // kv (8192 x 768)
  float* W1 = W0 + KVD;      // K proj (8192)
  float* W2 = W1 + KVD;      // V proj (8192)
  float* Av = W2 + KVD;      // state
  float* Bb = Av + RD;       // norm out
  float* Cb = Bb + RD;       // q / gate / scratch
  float* Db = Cb + RD;       // o / scratch
  float* Eb = Db + RD;       // attn out / gemm out
  float* Gb = Eb + RD;       // laned / mixer K
  float* Hq = Gb + RD;
  float* Hk = Hq + RD;
  float* Hv = Hk + RD;
  float* Ff = Hv + RD;       // 4096 x 3072
  float* s0 = Ff + (size_t)R_ROWS*D_FF;   // bpre
  float* s1 = s0 + (size_t)R_ROWS*N_HEAD; // gpre
  float* s2 = s1 + (size_t)R_ROWS*N_HEAD; // beta
  float* s3 = s2 + (size_t)R_ROWS*N_HEAD; // g

  auto gemm = [&](const float* A_, const bf16* Wm, float* C_, int M, int N, int K) {
    dim3 grid(M/128, (N + 127)/128);
    gemm_bf16_kernel<<<grid, dim3(256), 0, stream>>>(A_, Wm, C_, M, N, K);
  };
  auto rms = [&](const float* x, const float* w, float* y, int rows) {
    rmsnorm768_kernel<<<rows, 256, 0, stream>>>(x, w, y);
  };
  auto addv = [&](const float* a, const float* b, float* y, size_t n) {
    add_kernel<<<(int)((n + 255)/256), 256, 0, stream>>>(a, b, y, n);
  };
  const float scale = 0.125f;  // dh^-0.5

  // ================= fusion =================
  concat_kernel<<<KV_ROWS, 256, 0, stream>>>(precepts, prev_act, W0);
  rms(W0, f_norm_kv, W0, KV_ROWS);
  broadcast_kernel<<<R_ROWS, 256, 0, stream>>>(f_latents, Bb);   // Bb = q_in (residual)
  gemm(Bb, tf_wq, Cb, R_ROWS, D_MODEL, D_MODEL);
  gemm(W0, tf_wk, W1, KV_ROWS, D_MODEL, D_MODEL);
  gemm(W0, tf_wv, W2, KV_ROWS, D_MODEL, D_MODEL);
  attn_kernel<<<BT*N_HEAD, 32, 0, stream>>>(Cb, W1, W2, Eb, 32, scale);
  gemm(Eb, tf_wo, Db, R_ROWS, D_MODEL, D_MODEL);
  addv(Db, Bb, Av, RD);                                          // state = attn + q_in
  rms(Av, f_ffn_norm, Bb, R_ROWS);
  gemm(Bb, tf_w1, Ff, R_ROWS, D_FF, D_MODEL);
  gelu_kernel<<<(int)((RD*4 + 255)/256), 256, 0, stream>>>(Ff, (size_t)R_ROWS*D_FF);
  gemm(Ff, tf_w2, Eb, R_ROWS, D_MODEL, D_FF);
  addv(Av, Eb, Av, RD);

  // ================= memory blocks =================
  for (int b = 0; b < 4; ++b) {
    for (int l = 0; l < 2; ++l) {
      const LayerP& L = blk[b].lay[l];
      const LayW&  Lw = bw[b].lay[l];
      rms(Av, L.norm1, Bb, R_ROWS);
      gemm(Bb, Lw.wq, Cb, R_ROWS, D_MODEL, D_MODEL);
      gemm(Bb, Lw.wk, Gb, R_ROWS, D_MODEL, D_MODEL);
      gemm(Bb, Lw.wv, Hq, R_ROWS, D_MODEL, D_MODEL);
      attn_kernel<<<BT*N_HEAD, 32, 0, stream>>>(Cb, Gb, Hq, Eb, 16, scale);
      gemm(Eb, Lw.wo, Db, R_ROWS, D_MODEL, D_MODEL);
      addv(Av, Db, Av, RD);
      rms(Av, L.norm2, Bb, R_ROWS);
      gemm(Bb, Lw.w1, Ff, R_ROWS, D_FF, D_MODEL);
      gelu_kernel<<<(int)((RD*4 + 255)/256), 256, 0, stream>>>(Ff, (size_t)R_ROWS*D_FF);
      gemm(Ff, Lw.w2, Eb, R_ROWS, D_MODEL, D_FF);
      addv(Av, Eb, Av, RD);
    }
    const GdnP& G  = blk[b].gdn;
    const GdnW& Gw = bw[b].gdn;
    rms(Av, blk[b].gdn_norm, Bb, R_ROWS);
    perm_to_laned_kernel<<<R_ROWS, 256, 0, stream>>>(Bb, Gb);
    gemm(Gb, Gw.wq, Cb, R_ROWS, D_MODEL, D_MODEL);
    conv_silu_kernel<<<(int)((RD + 255)/256), 256, 0, stream>>>(Cb, G.cq, Hq);
    gemm(Gb, Gw.wk, Cb, R_ROWS, D_MODEL, D_MODEL);
    conv_silu_kernel<<<(int)((RD + 255)/256), 256, 0, stream>>>(Cb, G.ck, Hk);
    gemm(Gb, Gw.wv, Cb, R_ROWS, D_MODEL, D_MODEL);
    conv_silu_kernel<<<(int)((RD + 255)/256), 256, 0, stream>>>(Cb, G.cv, Hv);
    l2norm_kernel<<<(R_ROWS*N_HEAD)/8, 256, 0, stream>>>(Hq, 0.125f, R_ROWS*N_HEAD);
    l2norm_kernel<<<(R_ROWS*N_HEAD)/8, 256, 0, stream>>>(Hk, 1.0f,   R_ROWS*N_HEAD);
    gemm(Gb, Gw.wg, Cb, R_ROWS, D_MODEL, D_MODEL);             // gate
    gemm(Gb, Gw.wb, s0, R_ROWS, N_HEAD, D_MODEL);
    gemm(Gb, Gw.wa, s1, R_ROWS, N_HEAD, D_MODEL);
    beta_g_kernel<<<(R_ROWS*N_HEAD + 255)/256, 256, 0, stream>>>(
        s0, s1, G.A_log, G.dt_bias, s2, s3);
    gdn_scan_kernel<<<L_LANES*N_HEAD, 64, 0, stream>>>(Hq, Hk, Hv, s2, s3, Db);
    gated_norm_kernel<<<(R_ROWS*N_HEAD)/8, 256, 0, stream>>>(Db, Cb, G.norm_w, R_ROWS*N_HEAD);
    gemm(Db, Gw.wo, Eb, R_ROWS, D_MODEL, D_MODEL);
    perm_from_laned_add_kernel<<<R_ROWS, 256, 0, stream>>>(Av, Eb);
  }

  hipMemcpyAsync(d_out, Av, (size_t)out_size*sizeof(float),
                 hipMemcpyDeviceToDevice, stream);
}
